// LSTMModel_16406775070721
// MI455X (gfx1250) — matrix operand order repaired and checked by reading, not compiled, not run on hardware
//
#include <hip/hip_runtime.h>
#include <hip/hip_bf16.h>

// ---------------------------------------------------------------------------
// LSTMModel on MI455X (gfx1250):
//   BN folded into input weights; per-timestep GEMM [16,128]x[128,448] in
//   bf16 WMMA (v_wmma_f32_16x16x32_bf16), K = C(16)+H(112) = 128 exactly.
//   One WG = 16 batch rows, 7 waves; wave j owns hidden cols [16j,16j+16)
//   for all 4 gates -> gate nonlinearity + c/h update is wave-local.
//   Weights held in VGPRs (16 B-tiles/wave), h recirculated via LDS (bf16).
// ---------------------------------------------------------------------------

typedef __attribute__((ext_vector_type(16))) __bf16        v16bf;
typedef __attribute__((ext_vector_type(8)))  float         v8f;
typedef __attribute__((ext_vector_type(4)))  float         f32x4;
typedef __attribute__((ext_vector_type(4)))  unsigned int  u32x4;

#define BB 4096
#define TT 200
#define CC 16
#define HH 112
#define G4 448   // 4*HH

__device__ __forceinline__ unsigned short f2bf(float f) {
  unsigned int u = __float_as_uint(f);
  u += 0x7FFFu + ((u >> 16) & 1u);           // round-to-nearest-even
  return (unsigned short)(u >> 16);
}
__device__ __forceinline__ float fsig(float x) {
  return __builtin_amdgcn_rcpf(1.0f + __builtin_amdgcn_exp2f(-1.44269504f * x));
}
__device__ __forceinline__ float ftanh(float x) {
  return 2.0f * __builtin_amdgcn_rcpf(1.0f + __builtin_amdgcn_exp2f(-2.88539008f * x)) - 1.0f;
}

// ---------------------------------------------------------------------------
// Kernel 1: per-channel sum / sum-of-squares partials (deterministic, no
// float atomics). 256 blocks x 256 threads; thread's channel = tid%16 since
// all strides are multiples of 16. partial layout: [block][ {sum[16], sq[16]} ]
// ---------------------------------------------------------------------------
__global__ void bn_stats_kernel(const float* __restrict__ x, float* __restrict__ partial) {
  const int tid = threadIdx.x;
  float s = 0.f, s2 = 0.f;
  const size_t total = (size_t)BB * TT * CC;              // 13,107,200 = 65536*200
  for (size_t i = (size_t)blockIdx.x * 256 + tid; i < total; i += (size_t)256 * 256) {
    float v = x[i];
    s += v; s2 += v * v;
  }
  __shared__ float ls[256], lq[256];
  ls[tid] = s; lq[tid] = s2;
  __syncthreads();
  if (tid < 16) {
    float a = 0.f, b = 0.f;
    for (int g = 0; g < 16; ++g) { a += ls[g * 16 + tid]; b += lq[g * 16 + tid]; }
    partial[blockIdx.x * 32 + tid]      = a;
    partial[blockIdx.x * 32 + 16 + tid] = b;
  }
}

// ---------------------------------------------------------------------------
// Kernel 2: reduce stats, fold BN into W_ih, build effective bias, and
// pre-swizzle the [K=128][N=448] weight matrix into the exact WMMA B-matrix
// lane layout (bf16): Bw[ntile(28)][kblk(4)][lane(32)][e(16)], where
//   N = ntile*16 + lane%16,  K = kblk*32 + (lane>=16 ? 8 : 0) + (e&7) + (e>=8 ? 16 : 0).
// ---------------------------------------------------------------------------
__global__ void prep_kernel(const float* __restrict__ partial,
                            const float* __restrict__ gamma, const float* __restrict__ beta,
                            const float* __restrict__ W_ih,  const float* __restrict__ W_hh,
                            const float* __restrict__ b_ih,  const float* __restrict__ b_hh,
                            float* __restrict__ beff, unsigned short* __restrict__ Bw) {
  __shared__ float scale[16], shift[16];
  const int tid = threadIdx.x;
  if (tid < 16) {
    float s = 0.f, s2 = 0.f;
    for (int b = 0; b < 256; ++b) { s += partial[b * 32 + tid]; s2 += partial[b * 32 + 16 + tid]; }
    const float invN = 1.0f / ((float)BB * (float)TT);
    float mean = s * invN;
    float var  = s2 * invN - mean * mean;    // biased variance (matches jnp.var)
    float sc   = gamma[tid] * rsqrtf(var + 1e-5f);
    scale[tid] = sc;
    shift[tid] = beta[tid] - mean * sc;
  }
  __syncthreads();
  for (int g4 = tid; g4 < G4; g4 += 256) {
    float acc = b_ih[g4] + b_hh[g4];
    #pragma unroll
    for (int c = 0; c < CC; ++c) acc += W_ih[g4 * CC + c] * shift[c];
    beff[g4] = acc;
  }
  for (int idx = tid; idx < 28 * 4 * 32 * 16; idx += 256) {
    int e     = idx & 15;
    int lane  = (idx >> 4) & 31;
    int kblk  = (idx >> 9) & 3;
    int ntile = idx >> 11;
    int N = ntile * 16 + (lane & 15);
    int K = kblk * 32 + ((lane >> 4) << 3) + (e & 7) + ((e >> 3) << 4);
    float v = (K < CC) ? W_ih[N * CC + K] * scale[K] : W_hh[N * HH + (K - CC)];
    Bw[idx] = f2bf(v);
  }
}

// ---------------------------------------------------------------------------
// Kernel 3: persistent WMMA LSTM + FC. Grid: 256 WGs x 224 threads (7 waves).
// ---------------------------------------------------------------------------
__launch_bounds__(224)
__global__ void lstm_kernel(const float* __restrict__ x,
                            const float* __restrict__ beff,
                            const unsigned short* __restrict__ Bw,
                            const float* __restrict__ W_fc, const float* __restrict__ b_fc,
                            float* __restrict__ out) {
  __shared__ __align__(16) unsigned short hrow[16][HH];  // bf16 h, row-major
  __shared__ float hf[16][HH];                           // final f32 h for FC
  const int tid   = threadIdx.x;
  const int w     = tid >> 5;        // wave id = hidden tile j (0..6)
  const int lane  = tid & 31;
  const int nl    = lane & 15;       // N within tile / A-matrix row M
  const int lh    = lane >> 4;       // lane half (K sub-block select)
  const int bbase = blockIdx.x * 16;

  // --- Preload this wave's 16 weight B-tiles into registers (16 KB) ---
  v16bf Breg[4][4];
  #pragma unroll
  for (int g = 0; g < 4; ++g) {
    #pragma unroll
    for (int k = 0; k < 4; ++k) {
      const int ntile = g * 7 + w;
      Breg[g][k] = *(const v16bf*)(Bw + (size_t)((ntile * 4 + k) * 32 + lane) * 16);
    }
  }
  float bias[4];
  #pragma unroll
  for (int g = 0; g < 4; ++g) bias[g] = beff[g * HH + w * 16 + nl];

  // h0 = 0, c0 = 0
  for (int i = tid; i < 16 * HH; i += 224) ((unsigned short*)hrow)[i] = 0;
  v8f cacc = {0.f, 0.f, 0.f, 0.f, 0.f, 0.f, 0.f, 0.f};
  __syncthreads();

  // per-lane x pointer: row M=nl, channels lh*8 .. lh*8+7 (A kblk0, e0..7)
  const float* xrow = x + ((size_t)(bbase + nl) * TT) * CC + lh * 8;

  for (int t = 0; t < TT; ++t) {
    v8f z[4] = {
      {bias[0], bias[0], bias[0], bias[0], bias[0], bias[0], bias[0], bias[0]},
      {bias[1], bias[1], bias[1], bias[1], bias[1], bias[1], bias[1], bias[1]},
      {bias[2], bias[2], bias[2], bias[2], bias[2], bias[2], bias[2], bias[2]},
      {bias[3], bias[3], bias[3], bias[3], bias[3], bias[3], bias[3], bias[3]}};

    // K-blocks: build one A tile (WMMA A lane layout), feed all 4 gates.
    #pragma unroll
    for (int k = 0; k < 4; ++k) {
      union { v16bf v; unsigned short s[16]; u32x4 q[2]; } au;
      if (k == 0) {
        // e0..7: K = lh*8 + e  -> raw x channels (BN folded into weights)
        f32x4 x0 = *(const f32x4*)(xrow);
        f32x4 x1 = *(const f32x4*)(xrow + 4);
        au.s[0] = f2bf(x0.x); au.s[1] = f2bf(x0.y);
        au.s[2] = f2bf(x0.z); au.s[3] = f2bf(x0.w);
        au.s[4] = f2bf(x1.x); au.s[5] = f2bf(x1.y);
        au.s[6] = f2bf(x1.z); au.s[7] = f2bf(x1.w);
        // e8..15: K = 16 + lh*8 + (e-8) -> h[lh*8 .. +7]
        au.q[1] = *(const u32x4*)&hrow[nl][lh * 8];
      } else {
        const int n0 = k * 32 + lh * 8 - CC;   // K -> hidden index
        au.q[0] = *(const u32x4*)&hrow[nl][n0];
        au.q[1] = *(const u32x4*)&hrow[nl][n0 + 16];
      }
      #pragma unroll
      for (int g = 0; g < 4; ++g) {
        z[g] = __builtin_amdgcn_wmma_f32_16x16x32_bf16(
            false, au.v, false, Breg[g][k], (short)0, z[g], false, false);
      }
    }
    __syncthreads();  // all hrow reads for step t complete

    // Gate nonlinearities + state update (wave-local, D-tile layout:
    // VGPR r <-> row M = r + lh*8, col N = nl within hidden tile w).
    #pragma unroll
    for (int r = 0; r < 8; ++r) {
      float ig = fsig(z[0][r]);
      float fg = fsig(z[1][r]);
      float gg = ftanh(z[2][r]);
      float og = fsig(z[3][r]);
      float cn = fg * cacc[r] + ig * gg;
      cacc[r] = cn;
      float hv = og * ftanh(cn);
      hrow[r + lh * 8][w * 16 + nl] = f2bf(hv);
      if (t == TT - 1) hf[r + lh * 8][w * 16 + nl] = hv;
    }
    __syncthreads();  // h(t) visible before anyone builds A for t+1
    xrow += CC;
  }

  // --- FC: out[b][o] = tanh(h . W_fc[o] + b_fc[o]), 16 rows x 6 outputs ---
  if (tid < 96) {
    const int r = tid / 6, o = tid % 6;
    float acc = b_fc[o];
    for (int n = 0; n < HH; ++n) acc += hf[r][n] * W_fc[o * HH + n];
    out[(size_t)(bbase + r) * 6 + o] = tanhf(acc);
  }
}

// ---------------------------------------------------------------------------
extern "C" void kernel_launch(void* const* d_in, const int* in_sizes, int n_in,
                              void* d_out, int out_size, void* d_ws, size_t ws_size,
                              hipStream_t stream) {
  const float* x     = (const float*)d_in[0];
  const float* gamma = (const float*)d_in[1];
  const float* beta  = (const float*)d_in[2];
  const float* W_ih  = (const float*)d_in[3];
  const float* W_hh  = (const float*)d_in[4];
  const float* b_ih  = (const float*)d_in[5];
  const float* b_hh  = (const float*)d_in[6];
  const float* W_fc  = (const float*)d_in[7];
  const float* b_fc  = (const float*)d_in[8];
  float* out = (float*)d_out;

  char* ws = (char*)d_ws;
  float*          partial = (float*)ws;                    // 256*32 f32 = 32 KB
  float*          beff    = (float*)(ws + 32768);          // 448 f32
  unsigned short* Bw      = (unsigned short*)(ws + 40960); // 28*4*32*16 bf16 = 112 KB

  bn_stats_kernel<<<256, 256, 0, stream>>>(x, partial);
  prep_kernel<<<1, 256, 0, stream>>>(partial, gamma, beta, W_ih, W_hh, b_ih, b_hh, beff, Bw);
  lstm_kernel<<<256, 224, 0, stream>>>(x, beff, Bw, W_fc, b_fc, out);
}
